// AttributeBranch_46961172414953
// MI455X (gfx1250) — matrix operand order repaired and checked by reading, not compiled, not run on hardware
//
#include <hip/hip_runtime.h>
#include <hip/hip_bf16.h>

typedef __attribute__((ext_vector_type(16))) _Float16 v16h;
typedef __attribute__((ext_vector_type(8)))  _Float16 v8h;
typedef __attribute__((ext_vector_type(2)))  _Float16 v2h;
typedef __attribute__((ext_vector_type(8)))  float    v8f;

// ---------------------------------------------------------------------------
// Model constants
// ---------------------------------------------------------------------------
#define N_NODES   102400
#define N_EDGES   307200
#define N_DRUGS   256
#define NODES_PER_DRUG 400
#define N_CELL    256
#define DIM_DRUG  75
#define DIM_GEXP  697
#define DIM_METHY 808
#define DIM_FEAT  100
#define MUT_W     34673
#define CONV1_P   6795      // (34673-700)/5+1
#define POOL1_P   1359      // 6795/5
#define CONV2_P   678       // (1359-5)/2+1
#define FLAT_DIM  2010      // 30*67
#define KNN       6

// ---------------------------------------------------------------------------
// WMMA fragment helpers (ISA 7.12.2, wave32, 16-bit operands)
//  A 16xK: lane half h, row r; element e -> K = kc + ((e&8)?e+8:e) + 8*h
//    => two contiguous 8-half (16B) runs at kc+8h and kc+16+8h
//  B Kx16 (staged transposed BsT[n][k]): element e -> K = kc + ((e&8)?e+8:e) + 8*h
//    => two contiguous 8-half (16B) runs at kc+8h and kc+16+8h
//  C/D: VGPR i -> row = i + 8*h, col = lane&15
// ---------------------------------------------------------------------------
__device__ __forceinline__ v16h frag_a(const _Float16* rowp, int kc, int half) {
    v8h lo = *(const v8h*)(rowp + kc + 8 * half);
    v8h hi = *(const v8h*)(rowp + kc + 16 + 8 * half);
    v16h f;
#pragma unroll
    for (int e = 0; e < 8; ++e) { f[e] = lo[e]; f[e + 8] = hi[e]; }
    return f;
}
__device__ __forceinline__ v16h frag_b(const _Float16* rowp, int kc, int half) {
    v8h lo = *(const v8h*)(rowp + kc + 8 * half);
    v8h hi = *(const v8h*)(rowp + kc + 16 + 8 * half);
    v16h f;
#pragma unroll
    for (int e = 0; e < 8; ++e) { f[e] = lo[e]; f[e + 8] = hi[e]; }
    return f;
}

// ---------------------------------------------------------------------------
// Generic tiled GEMM:  C[M,N] = act(A[M,K] @ B[K,N] + bias)
// Block = 128 threads = 4 waves. Block tile: M=64, N=32, K-step 64.
// Each wave owns one 16-row M tile and both 16-col N tiles (A frag reused),
// issuing 4 v_wmma per staging barrier. grid = (ceil(N/32), ceil(M/64)).
// Staging has a block-uniform fast path (interior tiles, no guards) and a
// guarded path for boundary tiles only.
// act: 0 none, 1 relu, 2 tanh, 3 leaky(0.2)
// ---------------------------------------------------------------------------
__global__ void gemm_wmma_f16(const float* __restrict__ A,
                              const float* __restrict__ B,
                              const float* __restrict__ bias,
                              float* __restrict__ C,
                              int M, int N, int K, int ldc, int act) {
    __shared__ __align__(16) _Float16 As[64][72];   // 64 rows x 64 K (f16)
    __shared__ __align__(16) _Float16 BsT[32][72];  // 32 cols x 64 K (f16, transposed)

    const int tid  = threadIdx.x;
    const int wave = tid >> 5;
    const int lane = tid & 31;
    const int half = lane >> 4;
    const int r    = lane & 15;
    const int nBase = blockIdx.x * 32;
    const int mBase = blockIdx.y * 64;

    const bool fullM = (mBase + 64 <= M);
    const bool fullN = (nBase + 32 <= N);

    v8f acc0 = {};
    v8f acc1 = {};

    for (int k0 = 0; k0 < K; k0 += 64) {
        const bool fullK = (k0 + 64 <= K);

        // --- stage A 64x64 tile (f32 -> packed f16 pairs) ---
        if (fullM && fullK) {
            for (int l = tid; l < 64 * 32; l += 128) {
                int rr = l >> 5;
                int cp = (l & 31) * 2;
                const float* ap = A + (long)(mBase + rr) * K + k0 + cp;
                v2h p; p[0] = (_Float16)ap[0]; p[1] = (_Float16)ap[1];
                *(v2h*)&As[rr][cp] = p;
            }
        } else {
            for (int l = tid; l < 64 * 32; l += 128) {
                int rr = l >> 5;
                int cp = (l & 31) * 2;
                int gr = mBase + rr, gk = k0 + cp;
                const float* ap = A + (long)gr * K + gk;
                float f0 = (gr < M && gk     < K) ? ap[0] : 0.0f;
                float f1 = (gr < M && gk + 1 < K) ? ap[1] : 0.0f;
                v2h p; p[0] = (_Float16)f0; p[1] = (_Float16)f1;
                *(v2h*)&As[rr][cp] = p;
            }
        }
        // --- stage B 64x32 tile transposed (coalesced on N) ---
        if (fullN && fullK) {
            for (int l = tid; l < 64 * 32; l += 128) {
                int kk = l >> 5;
                int nn = l & 31;
                BsT[nn][kk] = (_Float16)B[(long)(k0 + kk) * N + nBase + nn];
            }
        } else {
            for (int l = tid; l < 64 * 32; l += 128) {
                int kk = l >> 5;
                int nn = l & 31;
                int gk = k0 + kk, gn = nBase + nn;
                float f0 = (gk < K && gn < N) ? B[(long)gk * N + gn] : 0.0f;
                BsT[nn][kk] = (_Float16)f0;
            }
        }
        // prefetch next A K-tile into cache while WMMAs run
        if (k0 + 64 < K) {
            int pr = mBase + (tid & 63);
            if (pr < M) __builtin_prefetch(&A[(long)pr * K + k0 + 64], 0, 0);
        }
        __syncthreads();

#pragma unroll
        for (int kc = 0; kc < 64; kc += 32) {
            v16h af = frag_a(&As[wave * 16 + r][0], kc, half);
            v16h b0 = frag_b(&BsT[r][0],      kc, half);
            v16h b1 = frag_b(&BsT[16 + r][0], kc, half);
            acc0 = __builtin_amdgcn_wmma_f32_16x16x32_f16(false, af, false, b0,
                                                          (short)0, acc0, false, false);
            acc1 = __builtin_amdgcn_wmma_f32_16x16x32_f16(false, af, false, b1,
                                                          (short)0, acc1, false, false);
        }
        __syncthreads();
    }

    const int col0 = nBase + r;
    const int col1 = nBase + 16 + r;
#pragma unroll
    for (int i = 0; i < 8; ++i) {
        int row = mBase + wave * 16 + i + 8 * half;
        if (row >= M) continue;
        float v0 = acc0[i], v1 = acc1[i];
        if (bias) {
            if (col0 < N) v0 += bias[col0];
            if (col1 < N) v1 += bias[col1];
        }
        if (act == 1)      { v0 = fmaxf(v0, 0.0f);               v1 = fmaxf(v1, 0.0f); }
        else if (act == 2) { v0 = tanhf(v0);                     v1 = tanhf(v1); }
        else if (act == 3) { v0 = (v0 >= 0.f) ? v0 : 0.2f * v0;  v1 = (v1 >= 0.f) ? v1 : 0.2f * v1; }
        if (col0 < N) C[(long)row * ldc + col0] = v0;
        if (col1 < N) C[(long)row * ldc + col1] = v1;
    }
}

// ---------------------------------------------------------------------------
// Mutation conv1 as implicit-im2col WMMA GEMM.
// X: (256, 34673), W: (50, 700) filter-major (already B-transposed layout),
// Y: (256, 50, 6795) = tanh(conv + bias).
// grid = (ceil(6795/16)=425, 256 cells), block = 128 (4 waves over 64 filters)
// ---------------------------------------------------------------------------
__global__ void conv1_wmma(const float* __restrict__ X,
                           const float* __restrict__ W,
                           const float* __restrict__ bias,
                           float* __restrict__ Y) {
    __shared__ __align__(16) _Float16 As[16][72];   // 16 positions x 64 K
    __shared__ __align__(16) _Float16 Ws[64][72];   // 64 filters   x 64 K (transposed B)

    const int tid  = threadIdx.x;
    const int wave = tid >> 5;
    const int lane = tid & 31;
    const int half = lane >> 4;
    const int r    = lane & 15;
    const int p0   = blockIdx.x * 16;
    const int cell = blockIdx.y;
    const long xb  = (long)cell * MUT_W;

    const bool fullP = (p0 + 16 <= CONV1_P);

    v8f acc = {};
    const int f = wave * 16 + r;   // output filter

    for (int k0 = 0; k0 < 700; k0 += 64) {
        const bool fullK = (k0 + 64 <= 700);

        // stage A: 16 positions x 64 taps (stride-5 windows, contiguous in k)
        if (fullP && fullK) {
            for (int l = tid; l < 16 * 32; l += 128) {
                int rr = l >> 5;
                int cp = (l & 31) * 2;
                const float* xp = X + xb + (long)(p0 + rr) * 5 + k0 + cp;
                v2h p; p[0] = (_Float16)xp[0]; p[1] = (_Float16)xp[1];
                *(v2h*)&As[rr][cp] = p;
            }
        } else {
            for (int l = tid; l < 16 * 32; l += 128) {
                int rr = l >> 5;
                int cp = (l & 31) * 2;
                int pos = p0 + rr, k = k0 + cp;
                const float* xp = X + xb + (long)pos * 5 + k;
                float f0 = (pos < CONV1_P && k     < 700) ? xp[0] : 0.0f;
                float f1 = (pos < CONV1_P && k + 1 < 700) ? xp[1] : 0.0f;
                v2h p; p[0] = (_Float16)f0; p[1] = (_Float16)f1;
                *(v2h*)&As[rr][cp] = p;
            }
        }
        // stage W: 64 filters x 64 taps; pair guard is exact (700 even, k even)
        for (int l = tid; l < 64 * 32; l += 128) {
            int ff = l >> 5;
            int cp = (l & 31) * 2;
            int k = k0 + cp;
            v2h p;
            if (ff < 50 && k < 700) {
                const float* wp = W + (long)ff * 700 + k;
                p[0] = (_Float16)wp[0]; p[1] = (_Float16)wp[1];
            } else {
                p[0] = (_Float16)0.0f;  p[1] = (_Float16)0.0f;
            }
            *(v2h*)&Ws[ff][cp] = p;
        }
        if (k0 + 64 < 700) {
            int pos = p0 + (tid >> 3);
            if (pos < CONV1_P)
                __builtin_prefetch(&X[xb + (long)pos * 5 + k0 + 64 + (tid & 7) * 8], 0, 0);
        }
        __syncthreads();

#pragma unroll
        for (int kc = 0; kc < 64; kc += 32) {
            v16h af = frag_a(&As[r][0], kc, half);
            v16h bf = frag_b(&Ws[wave * 16 + r][0], kc, half);
            acc = __builtin_amdgcn_wmma_f32_16x16x32_f16(false, af, false, bf,
                                                         (short)0, acc, false, false);
        }
        __syncthreads();
    }

    if (f < 50) {
        float bv = bias[f];
#pragma unroll
        for (int i = 0; i < 8; ++i) {
            int pos = p0 + i + 8 * half;
            if (pos < CONV1_P)
                Y[((long)cell * 50 + f) * CONV1_P + pos] = tanhf(acc[i] + bv);
        }
    }
}

// ---------------------------------------------------------------------------
// Small utility / irregular kernels
// ---------------------------------------------------------------------------
__global__ void fill_k(float* p, long n, float v) {
    long t = (long)blockIdx.x * 256 + threadIdx.x;
    if (t < n) p[t] = v;
}

__global__ void deg_edges_k(const int* __restrict__ tgt, float* __restrict__ deg, int ne) {
    int t = blockIdx.x * 256 + threadIdx.x;
    if (t < ne) atomicAdd(&deg[tgt[t]], 1.0f);
}

__global__ void deg_fin_k(const float* __restrict__ deg, float* __restrict__ dinv, int n) {
    int t = blockIdx.x * 256 + threadIdx.x;
    if (t < n) dinv[t] = rsqrtf(deg[t]);
}

// OUT[n,c] = H[n,c]/deg[n] + bias[c]   (self-loop term + bias, scatter target)
__global__ void gcn_init_k(const float* __restrict__ H, const float* __restrict__ bias,
                           const float* __restrict__ deg, float* __restrict__ OUT,
                           long total, int C) {
    long t = (long)blockIdx.x * 256 + threadIdx.x;
    if (t >= total) return;
    int c = (int)(t % C);
    long n = t / C;
    OUT[t] = H[t] / deg[n] + bias[c];
}

// one block per edge, thread per channel: OUT[tgt] += H[src]*dinv[tgt]*dinv[src]
__global__ void gcn_scatter_k(const float* __restrict__ H, const int* __restrict__ tgt,
                              const int* __restrict__ src, const float* __restrict__ dinv,
                              float* __restrict__ OUT, int C) {
    int c = threadIdx.x;
    if (c >= C) return;
    int e = blockIdx.x;
    int tn = tgt[e], sn = src[e];
    float w = dinv[tn] * dinv[sn];
    atomicAdd(&OUT[(long)tn * C + c], H[(long)sn * C + c] * w);
}

// per-column BN statistics over `rows`, optional relu folded in
__global__ void bn_stats_col(const float* __restrict__ X, int rows, int cols, int relu,
                             float* __restrict__ mean, float* __restrict__ istd) {
    __shared__ float ssum[256], ssq[256];
    int c = blockIdx.x;
    float s = 0.0f, q = 0.0f;
    for (int r = threadIdx.x; r < rows; r += 256) {
        float v = X[(long)r * cols + c];
        if (relu) v = fmaxf(v, 0.0f);
        s += v; q += v * v;
    }
    ssum[threadIdx.x] = s; ssq[threadIdx.x] = q;
    __syncthreads();
    for (int st = 128; st > 0; st >>= 1) {
        if (threadIdx.x < st) {
            ssum[threadIdx.x] += ssum[threadIdx.x + st];
            ssq[threadIdx.x]  += ssq[threadIdx.x + st];
        }
        __syncthreads();
    }
    if (threadIdx.x == 0) {
        float m = ssum[0] / rows;
        float v = ssq[0] / rows - m * m;
        mean[c] = m;
        istd[c] = rsqrtf(v + 1e-5f);
    }
}

__global__ void bn_apply_col(const float* __restrict__ X, float* __restrict__ Y,
                             long total, int cols, int relu,
                             const float* __restrict__ mean, const float* __restrict__ istd,
                             const float* __restrict__ g, const float* __restrict__ b) {
    long t = (long)blockIdx.x * 256 + threadIdx.x;
    if (t >= total) return;
    int c = (int)(t % cols);
    float v = X[t];
    if (relu) v = fmaxf(v, 0.0f);
    Y[t] = (v - mean[c]) * istd[c] * g[c] + b[c];
}

// BN2d stats over (cell, pos) for layout (cell, f, P)
__global__ void bn2d_stats(const float* __restrict__ X, int ncell, int nf, int P,
                           float* __restrict__ mean, float* __restrict__ istd) {
    __shared__ float ssum[256], ssq[256];
    int f = blockIdx.x;
    long n = (long)ncell * P;
    float s = 0.0f, q = 0.0f;
    for (long t = threadIdx.x; t < n; t += 256) {
        int cell = (int)(t / P);
        int pos  = (int)(t % P);
        float v = X[((long)cell * nf + f) * P + pos];
        s += v; q += v * v;
    }
    ssum[threadIdx.x] = s; ssq[threadIdx.x] = q;
    __syncthreads();
    for (int st = 128; st > 0; st >>= 1) {
        if (threadIdx.x < st) {
            ssum[threadIdx.x] += ssum[threadIdx.x + st];
            ssq[threadIdx.x]  += ssq[threadIdx.x + st];
        }
        __syncthreads();
    }
    if (threadIdx.x == 0) {
        float m = ssum[0] / (float)n;
        float v = ssq[0] / (float)n - m * m;
        mean[f] = m;
        istd[f] = rsqrtf(v + 1e-5f);
    }
}

// apply BN2d affine then max-pool (VALID, stride==window); output (cell, f, Pin/pool)
__global__ void bn2d_apply_pool(const float* __restrict__ X, float* __restrict__ Y,
                                int ncell, int nf, int Pin, int pool,
                                const float* __restrict__ mean, const float* __restrict__ istd,
                                const float* __restrict__ g, const float* __restrict__ b) {
    int Pout = Pin / pool;
    long total = (long)ncell * nf * Pout;
    long t = (long)blockIdx.x * 256 + threadIdx.x;
    if (t >= total) return;
    int q = (int)(t % Pout);
    long tf = t / Pout;
    int f = (int)(tf % nf);
    int cell = (int)(tf / nf);
    float sc = istd[f] * g[f];
    float sh = b[f] - mean[f] * sc;
    const float* xr = X + ((long)cell * nf + f) * Pin + q * pool;
    float m = -3.4e38f;
    for (int j = 0; j < pool; ++j) m = fmaxf(m, xr[j] * sc + sh);
    Y[t] = m;
}

// conv2: in (cell,50,1359), W (30,50,5) stride2, relu(+bias) -> (cell,30,678)
__global__ void conv2_k(const float* __restrict__ X, const float* __restrict__ W,
                        const float* __restrict__ bias, float* __restrict__ Y) {
    long total = (long)N_CELL * 30 * CONV2_P;
    long t = (long)blockIdx.x * 256 + threadIdx.x;
    if (t >= total) return;
    int p = (int)(t % CONV2_P);
    long tf = t / CONV2_P;
    int oc = (int)(tf % 30);
    int cell = (int)(tf / 30);
    float s = bias[oc];
    for (int ic = 0; ic < 50; ++ic) {
        const float* xr = X + ((long)cell * 50 + ic) * POOL1_P + p * 2;
        const float* wr = W + (oc * 50 + ic) * 5;
#pragma unroll
        for (int j = 0; j < 5; ++j) s += xr[j] * wr[j];
    }
    Y[t] = fmaxf(s, 0.0f);
}

// fused BN1d (stats+apply) for small row counts; block per column, safe in-place
__global__ void bn1d_fused(const float* __restrict__ X, float* __restrict__ Y,
                           int rows, int cols,
                           const float* __restrict__ g, const float* __restrict__ b) {
    __shared__ float ssum[256], ssq[256];
    __shared__ float sm, sis;
    int c = blockIdx.x;
    float s = 0.0f, q = 0.0f;
    for (int r = threadIdx.x; r < rows; r += 256) {
        float v = X[(long)r * cols + c];
        s += v; q += v * v;
    }
    ssum[threadIdx.x] = s; ssq[threadIdx.x] = q;
    __syncthreads();
    for (int st = 128; st > 0; st >>= 1) {
        if (threadIdx.x < st) {
            ssum[threadIdx.x] += ssum[threadIdx.x + st];
            ssq[threadIdx.x]  += ssq[threadIdx.x + st];
        }
        __syncthreads();
    }
    if (threadIdx.x == 0) {
        float m = ssum[0] / rows;
        sm = m;
        sis = rsqrtf(ssq[0] / rows - m * m + 1e-5f);
    }
    __syncthreads();
    float m = sm, is = sis, gc = g[c], bc = b[c];
    for (int r = threadIdx.x; r < rows; r += 256) {
        float v = X[(long)r * cols + c];
        Y[(long)r * cols + c] = (v - m) * is * gc + bc;
    }
}

// segment_max pooling over 400 nodes per drug -> Y[d*100+c]
__global__ void seg_max_k(const float* __restrict__ X, float* __restrict__ Y) {
    int t = blockIdx.x * 256 + threadIdx.x;
    if (t >= N_DRUGS * DIM_FEAT) return;
    int c = t % DIM_FEAT;
    int d = t / DIM_FEAT;
    const float* xr = X + (long)d * NODES_PER_DRUG * DIM_FEAT + c;
    float m = -3.4e38f;
    for (int j = 0; j < NODES_PER_DRUG; ++j) m = fmaxf(m, xr[(long)j * DIM_FEAT]);
    Y[t] = m;
}

// row L2 normalization; mode 0: x/max(||x||,1e-12), mode 1: x/(||x||+1e-10)
__global__ void rownorm_k(const float* __restrict__ X, int ldx,
                          float* __restrict__ Y, int ldy, int cols, int mode) {
    __shared__ float ss[128];
    __shared__ float sinv;
    int row = blockIdx.x;
    const float* xr = X + (long)row * ldx;
    float s = 0.0f;
    for (int c = threadIdx.x; c < cols; c += 128) { float v = xr[c]; s += v * v; }
    ss[threadIdx.x] = s;
    __syncthreads();
    for (int st = 64; st > 0; st >>= 1) {
        if (threadIdx.x < st) ss[threadIdx.x] += ss[threadIdx.x + st];
        __syncthreads();
    }
    if (threadIdx.x == 0) {
        float n = sqrtf(ss[0]);
        sinv = (mode == 0) ? 1.0f / fmaxf(n, 1e-12f) : 1.0f / (n + 1e-10f);
    }
    __syncthreads();
    float iv = sinv;
    for (int c = threadIdx.x; c < cols; c += 128)
        Y[(long)row * ldy + c] = xr[c] * iv;
}

// cosine-sim top-K per row from normalized 256x100 matrix
__global__ void sim_topk_k(const float* __restrict__ XN, int* __restrict__ idx) {
    __shared__ float s[N_CELL];
    int i = blockIdx.x, j = threadIdx.x;
    const float* xi = XN + (long)i * DIM_FEAT;
    const float* xj = XN + (long)j * DIM_FEAT;
    float d = 0.0f;
    for (int c = 0; c < DIM_FEAT; ++c) d += xi[c] * xj[c];
    s[j] = (j == i) ? 0.0f : d;
    __syncthreads();
    if (j == 0) {
        for (int kk = 0; kk < KNN; ++kk) {
            float best = -3.4e38f; int bi = 0;
            for (int t = 0; t < N_CELL; ++t)
                if (s[t] > best) { best = s[t]; bi = t; }
            idx[i * KNN + kk] = bi;
            s[bi] = -3.4e38f;
        }
    }
}

// sim-graph GCN aggregation: deg uniform = K+1 = 7; leaky(sum/7 + bias)
__global__ void simgcn_agg(const float* __restrict__ H, const int* __restrict__ idx,
                           const float* __restrict__ bias, float* __restrict__ Y, int C) {
    int t = blockIdx.x * 128 + threadIdx.x;
    if (t >= N_CELL * C) return;
    int c = t % C;
    int i = t / C;
    float s = H[(long)i * C + c];
    for (int kk = 0; kk < KNN; ++kk)
        s += H[(long)idx[i * KNN + kk] * C + c];
    s = s * (1.0f / 7.0f) + bias[c];
    Y[t] = (s >= 0.0f) ? s : 0.2f * s;
}

// attention fusion over 3 modality embeddings
__global__ void attn_k(const float* __restrict__ base, const float* __restrict__ m2,
                       const float* __restrict__ g2, const float* __restrict__ e2,
                       float* __restrict__ out) {
    __shared__ float p0[128], p1[128], p2[128];
    __shared__ float w0s, w1s, w2s;
    int i = blockIdx.x, c = threadIdx.x;
    float a = 0.0f, b = 0.0f, d = 0.0f;
    if (c < DIM_FEAT) {
        float bv = base[(long)i * DIM_FEAT + c];
        a = bv * m2[(long)i * DIM_FEAT + c];
        b = bv * g2[(long)i * DIM_FEAT + c];
        d = bv * e2[(long)i * DIM_FEAT + c];
    }
    p0[c] = a; p1[c] = b; p2[c] = d;
    __syncthreads();
    for (int st = 64; st > 0; st >>= 1) {
        if (c < st) { p0[c] += p0[c + st]; p1[c] += p1[c + st]; p2[c] += p2[c + st]; }
        __syncthreads();
    }
    if (c == 0) {
        float d0 = p0[0], d1 = p1[0], d2 = p2[0];
        float mx = fmaxf(d0, fmaxf(d1, d2));
        float e0 = expf(d0 - mx), e1 = expf(d1 - mx), e2v = expf(d2 - mx);
        float si = 1.0f / (e0 + e1 + e2v);
        w0s = e0 * si; w1s = e1 * si; w2s = e2v * si;
    }
    __syncthreads();
    if (c < DIM_FEAT)
        out[(long)i * DIM_FEAT + c] = w0s * m2[(long)i * DIM_FEAT + c]
                                    + w1s * g2[(long)i * DIM_FEAT + c]
                                    + w2s * e2[(long)i * DIM_FEAT + c];
}

// ---------------------------------------------------------------------------
// Host-side orchestration
// ---------------------------------------------------------------------------
extern "C" void kernel_launch(void* const* d_in, const int* in_sizes, int n_in,
                              void* d_out, int out_size, void* d_ws, size_t ws_size,
                              hipStream_t stream) {
    (void)in_sizes; (void)n_in; (void)out_size; (void)ws_size;

    // ---- inputs (setup_inputs dict order; params flattened in dict order) ----
    const float* drug_feature = (const float*)d_in[0];
    const int*   adj          = (const int*)d_in[1];
    // d_in[2] = drug_batch (uniform arange step 400) — pooling uses fixed stride
    const float* mut   = (const float*)d_in[3];
    const float* gexpr = (const float*)d_in[4];
    const float* methy = (const float*)d_in[5];
#define P(i) ((const float*)d_in[i])
    const float *Wg0=P(6),  *bg0=P(7),  *g0g=P(8),  *g0b=P(9);
    const float *Wg1=P(10), *bg1=P(11), *g1g=P(12), *g1b=P(13);
    const float *Wge=P(14), *bge=P(15), *geg=P(16), *geb=P(17);
    const float *Wx1=P(18), *bx1=P(19), *bnxg=P(20), *bnxb=P(21);
    const float *Wx2=P(22), *bx2=P(23);
    const float *Wm1=P(24), *bm1=P(25), *bnmg=P(26), *bnmb=P(27);
    const float *Wm2=P(28), *bm2=P(29);
    const float *Wc1=P(30), *bc1=P(31), *bn1g=P(32), *bn1b=P(33);
    const float *Wc2=P(34), *bc2=P(35), *bn2g=P(36), *bn2b=P(37);
    const float *bn3g=P(38), *bn3b=P(39);
    const float *Wmf=P(40), *bmf=P(41);
    const float *Wcb=P(42), *bcb=P(43);
    const float *Wsm0=P(44), *Wsm1=P(45), *bsm0=P(46), *bsm1=P(47);
    const float *Wsg0=P(48), *Wsg1=P(49), *bsg0=P(50), *bsg1=P(51);
    const float *Wse0=P(52), *Wse1=P(53), *bse0=P(54), *bse1=P(55);
    const float *Wcf=P(56), *bcf=P(57);
    const float *bag=P(58), *bab=P(59);
#undef P

    const int* src = adj;            // drug_adj[0]
    const int* tgt = adj + N_EDGES;  // drug_adj[1]

    // ---- workspace layout (float units) ----
    float* w = (float*)d_ws;
    const long SZ_NODE256 = (long)N_NODES * 256;
    long off = 0;
    float* X    = w + off; off += SZ_NODE256;                       // GCN layer input
    float* H    = w + off; off += SZ_NODE256;                       // x @ W
    float* OUT  = w + off; off += SZ_NODE256;                       // aggregated
    float* deg  = w + off; off += N_NODES;
    float* dinv = w + off; off += N_NODES;
    float* c1o  = w + off; off += (long)N_CELL * 50 * CONV1_P;      // conv1 out
    float* p1o  = w + off; off += (long)N_CELL * 50 * POOL1_P;      // pool1
    float* c2o  = w + off; off += (long)N_CELL * 30 * CONV2_P;      // conv2 out
    float* flat = w + off; off += (long)N_CELL * FLAT_DIM;          // (256,2010)
    float* cat  = w + off; off += (long)N_CELL * 300;               // [mut|gex|met]
    float* xgh  = w + off; off += (long)N_CELL * 256;
    float* xeh  = w + off; off += (long)N_CELL * 256;
    float* base = w + off; off += (long)N_CELL * DIM_FEAT;
    float* xn   = w + off; off += (long)N_CELL * DIM_FEAT;
    float* t1   = w + off; off += (long)N_CELL * DIM_FEAT;
    float* hb   = w + off; off += (long)N_CELL * DIM_FEAT;
    float* m2   = w + off; off += (long)N_CELL * DIM_FEAT;
    float* g2   = w + off; off += (long)N_CELL * DIM_FEAT;
    float* e2   = w + off; off += (long)N_CELL * DIM_FEAT;
    float* xcell= w + off; off += (long)N_CELL * DIM_FEAT;
    float* xall = w + off; off += 512L * DIM_FEAT;
    float* mean = w + off; off += 2048;
    float* istd = w + off; off += 2048;
    int*   knn  = (int*)(w + off); off += 3 * N_CELL * KNN;

    auto blocks = [](long n) { return (unsigned)((n + 255) / 256); };
    auto ggrid  = [](int M, int N) { return dim3((N + 31) / 32, (M + 63) / 64); };

    // =================== drug GCN branch ===================
    fill_k<<<blocks(N_NODES), 256, 0, stream>>>(deg, N_NODES, 1.0f);   // self-loop
    deg_edges_k<<<blocks(N_EDGES), 256, 0, stream>>>(tgt, deg, N_EDGES);
    deg_fin_k<<<blocks(N_NODES), 256, 0, stream>>>(deg, dinv, N_NODES);

    struct Layer { const float* Xin; const float* W; const float* b;
                   const float* g; const float* bb; int Kin; int Cout; };
    Layer layers[3] = {
        { drug_feature, Wg0, bg0, g0g, g0b, DIM_DRUG, 256 },
        { X,            Wg1, bg1, g1g, g1b, 256,      256 },
        { X,            Wge, bge, geg, geb, 256,      DIM_FEAT },
    };
    for (int li = 0; li < 3; ++li) {
        const Layer& L = layers[li];
        long total = (long)N_NODES * L.Cout;
        gemm_wmma_f16<<<ggrid(N_NODES, L.Cout), 128, 0, stream>>>(
            L.Xin, L.W, nullptr, H, N_NODES, L.Cout, L.Kin, L.Cout, 0);
        gcn_init_k<<<blocks(total), 256, 0, stream>>>(H, L.b, deg, OUT, total, L.Cout);
        gcn_scatter_k<<<N_EDGES, (L.Cout <= 128 ? 128 : 256), 0, stream>>>(
            H, tgt, src, dinv, OUT, L.Cout);
        bn_stats_col<<<L.Cout, 256, 0, stream>>>(OUT, N_NODES, L.Cout, 1, mean, istd);
        bn_apply_col<<<blocks(total), 256, 0, stream>>>(OUT, X, total, L.Cout, 1,
                                                        mean, istd, L.g, L.bb);
    }
    // segment_max over 400 nodes/drug -> rows 256..511 of x_all
    seg_max_k<<<blocks(N_DRUGS * DIM_FEAT), 256, 0, stream>>>(X, xall + 256L * DIM_FEAT);

    // =================== mutation conv branch ===================
    conv1_wmma<<<dim3((CONV1_P + 15) / 16, N_CELL), 128, 0, stream>>>(mut, Wc1, bc1, c1o);
    bn2d_stats<<<50, 256, 0, stream>>>(c1o, N_CELL, 50, CONV1_P, mean, istd);
    bn2d_apply_pool<<<blocks((long)N_CELL * 50 * POOL1_P), 256, 0, stream>>>(
        c1o, p1o, N_CELL, 50, CONV1_P, 5, mean, istd, bn1g, bn1b);
    conv2_k<<<blocks((long)N_CELL * 30 * CONV2_P), 256, 0, stream>>>(p1o, Wc2, bc2, c2o);
    bn2d_stats<<<30, 256, 0, stream>>>(c2o, N_CELL, 30, CONV2_P, mean, istd);
    bn2d_apply_pool<<<blocks((long)N_CELL * FLAT_DIM), 256, 0, stream>>>(
        c2o, flat, N_CELL, 30, CONV2_P, 10, mean, istd, bn2g, bn2b);
    bn1d_fused<<<FLAT_DIM, 256, 0, stream>>>(flat, flat, N_CELL, FLAT_DIM, bn3g, bn3b);
    // x_mut -> cat[:, 0:100]
    gemm_wmma_f16<<<ggrid(N_CELL, DIM_FEAT), 128, 0, stream>>>(
        flat, Wmf, bmf, cat, N_CELL, DIM_FEAT, FLAT_DIM, 300, 1);

    // =================== gexpr / methylation MLPs ===================
    gemm_wmma_f16<<<ggrid(N_CELL, 256), 128, 0, stream>>>(
        gexpr, Wx1, bx1, xgh, N_CELL, 256, DIM_GEXP, 256, 2);
    bn1d_fused<<<256, 256, 0, stream>>>(xgh, xgh, N_CELL, 256, bnxg, bnxb);
    gemm_wmma_f16<<<ggrid(N_CELL, DIM_FEAT), 128, 0, stream>>>(
        xgh, Wx2, bx2, cat + 100, N_CELL, DIM_FEAT, 256, 300, 1);

    gemm_wmma_f16<<<ggrid(N_CELL, 256), 128, 0, stream>>>(
        methy, Wm1, bm1, xeh, N_CELL, 256, DIM_METHY, 256, 2);
    bn1d_fused<<<256, 256, 0, stream>>>(xeh, xeh, N_CELL, 256, bnmg, bnmb);
    gemm_wmma_f16<<<ggrid(N_CELL, DIM_FEAT), 128, 0, stream>>>(
        xeh, Wm2, bm2, cat + 200, N_CELL, DIM_FEAT, 256, 300, 1);

    // =================== cell base + kNN sim graphs ===================
    gemm_wmma_f16<<<ggrid(N_CELL, DIM_FEAT), 128, 0, stream>>>(
        cat, Wcb, bcb, base, N_CELL, DIM_FEAT, 300, DIM_FEAT, 3);
    rownorm_k<<<N_CELL, 128, 0, stream>>>(base, DIM_FEAT, base, DIM_FEAT, DIM_FEAT, 0);

    // kNN edges per modality (columns of cat)
    for (int mi = 0; mi < 3; ++mi) {
        rownorm_k<<<N_CELL, 128, 0, stream>>>(cat + mi * DIM_FEAT, 300,
                                              xn, DIM_FEAT, DIM_FEAT, 1);
        sim_topk_k<<<N_CELL, N_CELL, 0, stream>>>(xn, knn + mi * N_CELL * KNN);
    }

    // per-modality 2-layer sim-GCN branches
    struct Br { const float* W0; const float* b0; const float* W1; const float* b1;
                const int* idx; float* out; };
    Br brs[3] = {
        { Wsm0, bsm0, Wsm1, bsm1, knn + 0 * N_CELL * KNN, m2 },
        { Wsg0, bsg0, Wsg1, bsg1, knn + 1 * N_CELL * KNN, g2 },
        { Wse0, bse0, Wse1, bse1, knn + 2 * N_CELL * KNN, e2 },
    };
    for (int bi = 0; bi < 3; ++bi) {
        const Br& B = brs[bi];
        gemm_wmma_f16<<<ggrid(N_CELL, DIM_FEAT), 128, 0, stream>>>(
            cat, B.W0, nullptr, t1, N_CELL, DIM_FEAT, 300, DIM_FEAT, 0);
        simgcn_agg<<<(N_CELL * DIM_FEAT + 127) / 128, 128, 0, stream>>>(t1, B.idx, B.b0,
                                                                        hb, DIM_FEAT);
        gemm_wmma_f16<<<ggrid(N_CELL, DIM_FEAT), 128, 0, stream>>>(
            hb, B.W1, nullptr, t1, N_CELL, DIM_FEAT, DIM_FEAT, DIM_FEAT, 0);
        simgcn_agg<<<(N_CELL * DIM_FEAT + 127) / 128, 128, 0, stream>>>(t1, B.idx, B.b1,
                                                                        B.out, DIM_FEAT);
        rownorm_k<<<N_CELL, 128, 0, stream>>>(B.out, DIM_FEAT, B.out, DIM_FEAT, DIM_FEAT, 0);
    }

    // =================== attention fusion + final ===================
    attn_k<<<N_CELL, 128, 0, stream>>>(base, m2, g2, e2, xcell);
    gemm_wmma_f16<<<ggrid(N_CELL, DIM_FEAT), 128, 0, stream>>>(
        xcell, Wcf, bcf, xall, N_CELL, DIM_FEAT, DIM_FEAT, DIM_FEAT, 1);
    // final BN over concatenated (512, 100) -> d_out
    bn1d_fused<<<DIM_FEAT, 256, 0, stream>>>(xall, (float*)d_out, 512, DIM_FEAT, bag, bab);
}
